// Decoder_13657996001434
// MI455X (gfx1250) — matrix operand order repaired and checked by reading, not compiled, not run on hardware
//
#include <hip/hip_runtime.h>
#include <hip/hip_bf16.h>
#include <math.h>

// Problem constants
#define B_   32
#define S_   2048
#define F_   512
#define H_   512
#define V_   1024
#define G_   2048   // 4*H
#define K1_  1024   // F+H
#define NWG  32
#define NTHR 256

typedef __attribute__((ext_vector_type(16))) __bf16 v16bf;
typedef __attribute__((ext_vector_type(8)))  float  v8f;

union ABfrag { v16bf v; uint4 u[2]; };

__device__ __forceinline__ unsigned short f2bf(float x) {
  unsigned u = __float_as_uint(x);
  u += 0x7FFFu + ((u >> 16) & 1u);            // round-to-nearest-even
  return (unsigned short)(u >> 16);
}
__device__ __forceinline__ float bflo(unsigned w){ return __uint_as_float(w << 16); }
__device__ __forceinline__ float bfhi(unsigned w){ return __uint_as_float(w & 0xFFFF0000u); }
__device__ __forceinline__ float sigm(float x){ return 1.0f / (1.0f + __expf(-x)); }

// ---- device-wide monotonic barrier (counter zeroed by hipMemsetAsync) ----
__device__ __forceinline__ void gsync(unsigned* bar, unsigned target) {
  __syncthreads();
  if (threadIdx.x == 0) {
    __threadfence();
    atomicAdd(bar, 1u);
    volatile unsigned* vb = (volatile unsigned*)bar;
    while (*vb < target) __builtin_amdgcn_s_sleep(2);
    __threadfence();
  }
  __syncthreads();
}

// ---- bulk-stage a 16 x Kc bf16 block (row stride srcStride) into LDS ----
// All threads of the WG participate; coalesced b128 loads, b128 LDS stores.
__device__ __forceinline__ void stage16(const unsigned short* __restrict__ src,
                                        int srcStride, unsigned short* dst,
                                        int Kc, int tid) {
  __syncthreads();                      // prior phase may still read this LDS
  const int vper = Kc / 8;              // uint4 vectors per row
  const int nvec = 16 * vper;
  for (int i = tid; i < nvec; i += NTHR) {
    int r = i / vper, cc = i - r * vper;
    *(uint4*)(dst + i * 8) = *(const uint4*)(src + (size_t)r * srcStride + cc * 8);
  }
  __syncthreads();
}

// ---- one 16x16 tile of A[16,K] (LDS) x B[16,K]^T (global), bf16/f32 acc ----
// A-frag (ISA 7.12.2, 16-bit A 16x32): lane half selects K 0-7/16-23 vs
// 8-15/24-31. B-frag (32x16): same K split as A.
__device__ __forceinline__ v8f mm_tile_lds(const unsigned short* ar,   // LDS row
                                           const unsigned short* br,   // global row
                                           int K, int khalf) {
  v8f c = {0.f,0.f,0.f,0.f,0.f,0.f,0.f,0.f};
  for (int k0 = 0; k0 < K; k0 += 32) {
    ABfrag a, b;
    a.u[0] = *(const uint4*)(ar + k0 + khalf * 8);        // regs 0-3
    a.u[1] = *(const uint4*)(ar + k0 + khalf * 8 + 16);   // regs 4-7
    b.u[0] = *(const uint4*)(br + k0 + khalf * 8);        // K 0-7 / 8-15
    b.u[1] = *(const uint4*)(br + k0 + khalf * 8 + 16);   // K 16-23 / 24-31
    c = __builtin_amdgcn_wmma_f32_16x16x32_bf16(false, a.v, false, b.v,
                                                (short)0, c, false, false);
  }
  return c;
}

__device__ __forceinline__ float block_reduce_max(float v, float* red) {
  int tid = threadIdx.x;
  red[tid] = v; __syncthreads();
  for (int s = NTHR / 2; s > 0; s >>= 1) {
    if (tid < s) red[tid] = fmaxf(red[tid], red[tid + s]);
    __syncthreads();
  }
  float r = red[0]; __syncthreads();
  return r;
}
__device__ __forceinline__ float block_reduce_sum(float v, float* red) {
  int tid = threadIdx.x;
  red[tid] = v; __syncthreads();
  for (int s = NTHR / 2; s > 0; s >>= 1) {
    if (tid < s) red[tid] = red[tid] + red[tid + s];
    __syncthreads();
  }
  float r = red[0]; __syncthreads();
  return r;
}

// =========================== init kernels ===========================
__global__ void k_conv_bf16(const float* __restrict__ src,
                            unsigned short* __restrict__ dst, size_t n) {
  size_t i = (size_t)blockIdx.x * blockDim.x + threadIdx.x;
  size_t st = (size_t)gridDim.x * blockDim.x;
  for (; i < n; i += st) dst[i] = f2bf(src[i]);
}

// Wg[n, 0:512] = W_ih[n,:], Wg[n, 512:1024] = W_hh[n,:]   (bf16, row-major)
__global__ void k_pack_gates_w(const float* __restrict__ wih,
                               const float* __restrict__ whh,
                               unsigned short* __restrict__ wg) {
  size_t i = (size_t)blockIdx.x * blockDim.x + threadIdx.x;
  size_t st = (size_t)gridDim.x * blockDim.x;
  for (; i < (size_t)G_ * F_; i += st) {
    size_t n = i / F_, k = i % F_;
    wg[n * K1_ + k]      = f2bf(wih[i]);
    wg[n * K1_ + F_ + k] = f2bf(whh[i]);
  }
}

__global__ void k_init_state(const float* __restrict__ features,
                             const float* __restrict__ h0,
                             const float* __restrict__ c0,
                             const float* __restrict__ b_ih,
                             const float* __restrict__ b_hh,
                             unsigned short* __restrict__ act,
                             float* __restrict__ cstate,
                             float* __restrict__ bsum) {
  int u = blockIdx.x * blockDim.x + threadIdx.x;     // 16384 threads
  if (u < B_ * H_) {
    int b = u >> 9, j = u & 511;
    act[b * K1_ + j]      = f2bf(features[(size_t)b * S_ * F_ + j]); // feat_0
    act[b * K1_ + F_ + j] = f2bf(h0[u]);                             // h_0
    cstate[u] = c0[u];
  }
  if (u < G_) bsum[u] = b_ih[u] + b_hh[u];
}

// ======================= persistent decoder =======================
__global__ void __launch_bounds__(NTHR)
decoder_persistent(const float* __restrict__ features,
                   const unsigned short* __restrict__ encb,   // [B,S,H] bf16
                   const unsigned short* __restrict__ Wg,     // [2048,1024] bf16
                   const unsigned short* __restrict__ Wqb,    // [512,512]  bf16
                   const unsigned short* __restrict__ Wcb,    // [1024,512] bf16
                   const float* __restrict__ bsum,            // [2048]
                   const float* __restrict__ bc,              // [1024]
                   unsigned short* act,                       // [32,1024] bf16
                   float* cstate,                             // [32,512]
                   float* gates,                              // [32,2048]
                   float* qbuf,                               // [32,512]
                   unsigned short* avb,                       // [32,512] bf16
                   float* __restrict__ out,                   // [32,2048,1024]
                   unsigned* bar) {
  const int tid   = threadIdx.x;
  const int wg    = blockIdx.x;
  const int lane  = tid & 31;
  const int wv    = tid >> 5;
  const int l15   = lane & 15;
  const int khalf = lane >> 4;
  const float scale = 0.044194173824159216f;   // 1/sqrt(512)

  // 32 KB staging buffer unioned with attention scratch (disjoint phases,
  // every transition is fenced by gsync/__syncthreads).
  __shared__ union {
    unsigned short actl[16 * K1_];                       // 32 KB A-operand
    struct { float sm[S_]; float qs[H_]; float red[NTHR]; } at;  // 11 KB
  } shu;

  unsigned bt = 0;

  for (int t = 0; t < S_; ++t) {
    // -- phase A: gates = [feat|h] @ [W_ih|W_hh]^T   (M32 N2048 K1024)
    // Each WG owns one M-tile (mt = wg>>4); its 8 waves cover 8 N-tiles.
    {
      int mt = wg >> 4;                 // 0..1
      int nt = (wg & 15) * 8 + wv;      // 0..127
      stage16(act + (size_t)mt * 16 * K1_, K1_, shu.actl, K1_, tid);
      const unsigned short* ar = shu.actl + l15 * K1_;
      const unsigned short* br = Wg + (size_t)(nt * 16 + l15) * K1_;
      v8f c = mm_tile_lds(ar, br, K1_, khalf);
      float* go = gates + (mt * 16 + khalf * 8) * G_ + nt * 16 + l15;
      #pragma unroll
      for (int r = 0; r < 8; ++r) go[r * G_] = c[r];   // C-layout: VGPR r -> row
    }
    bt += NWG; gsync(bar, bt);

    // -- phase B: LSTM pointwise; stage h_new (bf16) + next feat into act
    {
      int g0 = (wg * NTHR + tid) * 2;
      #pragma unroll
      for (int e = 0; e < 2; ++e) {
        int u = g0 + e;
        int b = u >> 9, j = u & 511;
        const float* gr = gates + b * G_;
        float iv = gr[j]          + bsum[j];
        float fv = gr[512 + j]    + bsum[512 + j];
        float gv = gr[1024 + j]   + bsum[1024 + j];
        float ov = gr[1536 + j]   + bsum[1536 + j];
        float cn = sigm(fv) * cstate[b * H_ + j] + sigm(iv) * tanhf(gv);
        float hn = sigm(ov) * tanhf(cn);
        cstate[b * H_ + j]    = cn;
        act[b * K1_ + F_ + j] = f2bf(hn);
        if (t + 1 < S_)
          act[b * K1_ + j] = f2bf(features[((size_t)b * S_ + t + 1) * F_ + j]);
      }
    }
    bt += NWG; gsync(bar, bt);

    // -- phase C: q = h_new @ Wq^T  (M32 N512 K512): WGs 0..7, 8 tiles each
    if (wg < 8) {
      int mt = wg >> 2;                 // 0..1
      int nt = (wg & 3) * 8 + wv;       // 0..31
      stage16(act + (size_t)mt * 16 * K1_ + F_, K1_, shu.actl, H_, tid);
      const unsigned short* ar = shu.actl + l15 * H_;
      const unsigned short* br = Wqb + (nt * 16 + l15) * H_;
      v8f c = mm_tile_lds(ar, br, H_, khalf);
      float* qo = qbuf + (mt * 16 + khalf * 8) * H_ + nt * 16 + l15;
      #pragma unroll
      for (int r = 0; r < 8; ++r) qo[r * H_] = c[r];
    }
    bt += NWG; gsync(bar, bt);

    // -- phase D: causal attention, one WG per batch row
    {
      int b = wg;
      const unsigned* erow = (const unsigned*)(encb + (size_t)b * S_ * H_);
      for (int k = tid; k < H_; k += NTHR) shu.at.qs[k] = qbuf[b * H_ + k];
      __syncthreads();
      float lmax = -3.4e38f;
      for (int p = tid; p <= t; p += NTHR) {          // scores
        const unsigned* e = erow + (size_t)p * (H_ / 2);
        float acc = 0.f;
        #pragma unroll 8
        for (int kk = 0; kk < H_ / 2; ++kk) {
          unsigned w = e[kk];
          acc += bflo(w) * shu.at.qs[2 * kk] + bfhi(w) * shu.at.qs[2 * kk + 1];
        }
        float sc = acc * scale;
        shu.at.sm[p] = sc;
        lmax = fmaxf(lmax, sc);
      }
      float mx = block_reduce_max(lmax, shu.at.red);
      float lsum = 0.f;
      for (int p = tid; p <= t; p += NTHR) {          // exp + sum
        float w = __expf(shu.at.sm[p] - mx);
        shu.at.sm[p] = w;
        lsum += w;
      }
      float tot = block_reduce_sum(lsum, shu.at.red);
      float inv = 1.0f / tot;
      // weighted sum: thread owns channels (2*tid, 2*tid+1); coalesced rows
      float a0 = 0.f, a1 = 0.f;
      for (int p = 0; p <= t; ++p) {
        float w = shu.at.sm[p];
        unsigned ee = erow[(size_t)p * (H_ / 2) + tid];
        a0 += w * bflo(ee);
        a1 += w * bfhi(ee);
      }
      avb[b * H_ + 2 * tid]     = f2bf(a0 * inv);
      avb[b * H_ + 2 * tid + 1] = f2bf(a1 * inv);
    }
    bt += NWG; gsync(bar, bt);

    // -- phase E: logits = av @ Wc^T + bc (M32 N1024 K512): WGs 0..15
    if (wg < 16) {
      int mt = wg >> 3;                 // 0..1
      int nt = (wg & 7) * 8 + wv;       // 0..63
      stage16(avb + (size_t)mt * 16 * H_, H_, shu.actl, H_, tid);
      const unsigned short* ar = shu.actl + l15 * H_;
      const unsigned short* br = Wcb + (size_t)(nt * 16 + l15) * H_;
      v8f c = mm_tile_lds(ar, br, H_, khalf);
      int col = nt * 16 + l15;
      float bias = bc[col];
      int row0 = mt * 16 + khalf * 8;
      #pragma unroll
      for (int r = 0; r < 8; ++r)
        out[((size_t)(row0 + r) * S_ + t) * V_ + col] = c[r] + bias;
    }
    // no sync needed: next iteration's first barrier orders phase E vs t+1,
    // and stage16() begins with __syncthreads for intra-WG LDS reuse.
  }
}

// =========================== host launcher ===========================
extern "C" void kernel_launch(void* const* d_in, const int* in_sizes, int n_in,
                              void* d_out, int out_size, void* d_ws, size_t ws_size,
                              hipStream_t stream) {
  (void)in_sizes; (void)n_in; (void)out_size;
  const float* features  = (const float*)d_in[0];   // [B,S,F]
  const float* encodings = (const float*)d_in[1];   // [B,S,H]
  const float* h0        = (const float*)d_in[2];   // [B,H]
  const float* c0        = (const float*)d_in[3];   // [B,H]
  const float* W_ih      = (const float*)d_in[4];   // [4H,F]
  const float* W_hh      = (const float*)d_in[5];   // [4H,H]
  const float* b_ih      = (const float*)d_in[6];   // [4H]
  const float* b_hh      = (const float*)d_in[7];   // [4H]
  const float* Wq        = (const float*)d_in[8];   // [H,H]
  const float* Wc        = (const float*)d_in[9];   // [V,H]
  const float* bc        = (const float*)d_in[10];  // [V]
  float* out = (float*)d_out;

  // workspace carve-up (~70 MB total)
  char* ws = (char*)d_ws;
  size_t off = 0;
  auto alloc = [&](size_t bytes) -> char* {
    char* p = ws + off;
    off += (bytes + 255) & ~(size_t)255;
    return p;
  };
  unsigned*       bar    = (unsigned*)      alloc(256);
  unsigned short* encb   = (unsigned short*)alloc((size_t)B_ * S_ * H_ * 2);  // 64 MB
  unsigned short* Wg     = (unsigned short*)alloc((size_t)G_ * K1_ * 2);      // 4 MB
  unsigned short* Wqb    = (unsigned short*)alloc((size_t)H_ * H_ * 2);
  unsigned short* Wcb    = (unsigned short*)alloc((size_t)V_ * H_ * 2);
  float*          bsum   = (float*)         alloc((size_t)G_ * 4);
  unsigned short* act    = (unsigned short*)alloc((size_t)B_ * K1_ * 2);
  float*          cstate = (float*)         alloc((size_t)B_ * H_ * 4);
  float*          gates  = (float*)         alloc((size_t)B_ * G_ * 4);
  float*          qbuf   = (float*)         alloc((size_t)B_ * H_ * 4);
  unsigned short* avb    = (unsigned short*)alloc((size_t)B_ * H_ * 2);
  (void)ws_size;

  hipMemsetAsync(bar, 0, 256, stream);

  k_conv_bf16<<<4096, 256, 0, stream>>>(encodings, encb, (size_t)B_ * S_ * H_);
  k_conv_bf16<<<512,  256, 0, stream>>>(Wq, Wqb, (size_t)H_ * H_);
  k_conv_bf16<<<512,  256, 0, stream>>>(Wc, Wcb, (size_t)V_ * H_);
  k_pack_gates_w<<<1024, 256, 0, stream>>>(W_ih, W_hh, Wg);
  k_init_state<<<(B_ * H_ + 255) / 256, 256, 0, stream>>>(
      features, h0, c0, b_ih, b_hh, act, cstate, bsum);

  decoder_persistent<<<NWG, NTHR, 0, stream>>>(
      features, encb, Wg, Wqb, Wcb, bsum, bc,
      act, cstate, gates, qbuf, avb, out, bar);
}
